// MultiheadAttention_50611894616492
// MI455X (gfx1250) — matrix operand order repaired and checked by reading, not compiled, not run on hardware
//
#include <hip/hip_runtime.h>

#define BB 8
#define NN 2048
#define DD 1024
#define SS (BB*NN)           // 16384 total rows
#define NEG_INF (-3.0e38f)

typedef unsigned short bf16r;   // raw bf16 storage
typedef __attribute__((ext_vector_type(16))) __bf16 v16bf;
typedef __attribute__((ext_vector_type(8)))  float   v8f;

union V16 { v16bf v; uint4 u[2]; };
union BF4 { bf16r h[4]; ushort4 u; };
union BF8 { bf16r h[8]; uint4 u; };

#define WMMA_BF16(a,b,c) __builtin_amdgcn_wmma_f32_16x16x32_bf16(false,(a),false,(b),(short)0,(c),false,false)

// float -> bf16 round-to-nearest-even (bit manipulation, host-parse safe)
__device__ __forceinline__ bf16r f2bf(float f) {
    unsigned u = __float_as_uint(f);
    u += 0x7fffu + ((u >> 16) & 1u);
    return (bf16r)(u >> 16);
}

// ---- WMMA operand loaders (ISA 7.12.2 layouts) ----
// A 16x32 bf16: lane = row(M)%16, half h = lane>>4.
//   h=0: VGPR0-3 <- K0..7,  VGPR4-7 <- K16..23
//   h=1: VGPR0-3 <- K8..15, VGPR4-7 <- K24..31
__device__ __forceinline__ v16bf load_A(const bf16r* base, int ld) {
    const int lane = threadIdx.x & 31;
    const int row  = lane & 15;
    const int h    = lane >> 4;
    const bf16r* p = base + (size_t)row * ld + h * 8;
    V16 r;
    r.u[0] = *reinterpret_cast<const uint4*>(p);
    r.u[1] = *reinterpret_cast<const uint4*>(p + 16);
    return r.v;
}

// B 32x16 bf16: lane = col(N)%16, same K layout as A (h=0: K0..7,K16..23; h=1: K8..15,K24..31).
__device__ __forceinline__ v16bf load_B(const bf16r* base, int ld) {
    const int lane = threadIdx.x & 31;
    const int col  = lane & 15;
    const int h    = lane >> 4;
    const bf16r* p = base + (size_t)col * ld + h * 8;
    V16 r;
    r.u[0] = *reinterpret_cast<const uint4*>(p);
    r.u[1] = *reinterpret_cast<const uint4*>(p + 16);
    return r.v;
}

// ---------------- Kernel 1a: fp32 -> bf16 convert of x ----------------
__global__ void k_cvt_x(const float* __restrict__ x, bf16r* __restrict__ xh, int n4) {
    int i = blockIdx.x * blockDim.x + threadIdx.x;
    if (i < n4) {
        float4 f = reinterpret_cast<const float4*>(x)[i];
        BF4 o;
        o.h[0] = f2bf(f.x); o.h[1] = f2bf(f.y); o.h[2] = f2bf(f.z); o.h[3] = f2bf(f.w);
        reinterpret_cast<ushort4*>(xh)[i] = o.u;
    }
}

// ---------------- Kernel 1b: convert + transpose weight (D x D) ----------------
__global__ void k_cvt_wT(const float* __restrict__ W, bf16r* __restrict__ WT) {
    int t = blockIdx.x * blockDim.x + threadIdx.x;   // over D*D
    int k = t & (DD - 1);
    int n = t >> 10;
    WT[(size_t)n * DD + k] = f2bf(W[(size_t)k * DD + n]);
}

// ---------------- Kernel 2: fused QKV projection ----------------
// Wave: 32 rows x 64 cols for Q,K,V. B stream software-pipelined (bcur/bnxt).
__global__ __launch_bounds__(128) void k_qkv(
    const bf16r* __restrict__ xh,
    const bf16r* __restrict__ wqT, const bf16r* __restrict__ wkT, const bf16r* __restrict__ wvT,
    const float* __restrict__ bq,  const float* __restrict__ bk,  const float* __restrict__ bv,
    bf16r* __restrict__ qh, bf16r* __restrict__ kh, bf16r* __restrict__ vT)
{
    const int wave = threadIdx.x >> 5;
    const int lane = threadIdx.x & 31;
    const int m0 = (blockIdx.x * 4 + wave) * 32;     // 32-row tile
    const int n0 = blockIdx.y * 64;

    const bf16r* wptr[3] = { wqT, wkT, wvT };
    const bf16r* wb[12];
#pragma unroll
    for (int s = 0; s < 12; ++s)
        wb[s] = wptr[s >> 2] + (size_t)(n0 + 16 * (s & 3)) * DD;

    v8f acc[12][2] = {};                 // [w*4+j][mt]
    v16bf bcur = load_B(wb[0], DD);      // prime the pipeline
    for (int kb = 0; kb < DD; kb += 32) {
        v16bf a0 = load_A(xh + (size_t)m0 * DD + kb, DD);
        v16bf a1 = load_A(xh + (size_t)(m0 + 16) * DD + kb, DD);
        if (kb + 32 < DD) {
            __builtin_prefetch(xh + (size_t)m0 * DD + kb + 32, 0, 3);
            __builtin_prefetch(xh + (size_t)(m0 + 16) * DD + kb + 32, 0, 3);
        }
        const int kbn = (kb + 32 < DD) ? kb + 32 : 0;  // next k-block (wraps harmlessly)
#pragma unroll
        for (int s = 0; s < 12; ++s) {
            const int sn = (s + 1) % 12;
            v16bf bn = load_B(wb[sn] + ((s == 11) ? kbn : kb), DD);
            acc[s][0] = WMMA_BF16(a0, bcur, acc[s][0]);
            acc[s][1] = WMMA_BF16(a1, bcur, acc[s][1]);
            bcur = bn;
        }
    }
    const int h  = lane >> 4;
    const int cl = lane & 15;
    const float qscale = 0.03125f * 1.44269504088896340736f;  // (1/sqrt(1024))*log2(e)
#pragma unroll
    for (int j = 0; j < 4; ++j) {
        const int col = n0 + 16 * j + cl;
        const float biasq = bq[col], biask = bk[col], biasv = bv[col];
#pragma unroll
        for (int mt = 0; mt < 2; ++mt) {
            // Q / K: row-major, lanes coalesce along columns
#pragma unroll
            for (int r = 0; r < 8; ++r) {
                const int row = m0 + mt * 16 + r + 8 * h;
                qh[(size_t)row * DD + col] = f2bf((acc[j][mt][r] + biasq) * qscale);
                kh[(size_t)row * DD + col] = f2bf(acc[4 + j][mt][r] + biask);
            }
            // V transposed: lane holds 8 consecutive rows of its column -> one b128 store
            BF8 pk;
#pragma unroll
            for (int r = 0; r < 8; ++r) pk.h[r] = f2bf(acc[8 + j][mt][r] + biasv);
            const int row0 = m0 + mt * 16 + 8 * h;       // 8-row chunk start
            const int bidx = row0 >> 11;
            const int rin  = row0 & (NN - 1);
            *reinterpret_cast<uint4*>(&vT[((size_t)bidx * DD + col) * NN + rin]) = pk.u;
        }
    }
}

// ---------------- Kernel 3: flash attention ----------------
__global__ __launch_bounds__(256) void k_attn(
    const bf16r* __restrict__ qh, const bf16r* __restrict__ kh,
    const bf16r* __restrict__ vT, bf16r* __restrict__ oh)
{
    __shared__ alignas(16) bf16r Qs[16 * DD];      // 32 KB query tile
    __shared__ alignas(16) float Ss[16 * 128];     // 8 KB score block
    __shared__ alignas(16) bf16r Ps[16 * 128];     // 4 KB prob block (bf16)
    __shared__ float pm[16][16], pl[16][16];
    __shared__ float m_run[16], l_run[16], m_new[16], corr[16];

    const int t    = threadIdx.x;
    const int wave = t >> 5;
    const int lane = t & 31;
    const int h    = lane >> 4;
    const int cl   = lane & 15;
    const int bidx = blockIdx.x >> 7;              // batch
    const int qt   = blockIdx.x & 127;             // query tile in batch
    const int m0   = bidx * NN + qt * 16;          // global row base
    const bf16r* kbase = kh + (size_t)bidx * NN * DD;
    const bf16r* vtb   = vT + (size_t)bidx * DD * NN;

    {   // stage Q tile (coalesced b128)
        const uint4* src = reinterpret_cast<const uint4*>(qh + (size_t)m0 * DD);
        uint4* dst = reinterpret_cast<uint4*>(Qs);
#pragma unroll
        for (int i = 0; i < 8; ++i) dst[t + 256 * i] = src[t + 256 * i];
    }
    if (t < 16) { m_run[t] = NEG_INF; l_run[t] = 0.0f; }
    v8f o[8] = {};
    __syncthreads();

    for (int it = 0; it < 16; ++it) {
        const int key0 = it * 128;
        // --- scores: keys [key0+wave*16, +16) over full D, B stream pipelined
        {
            const bf16r* kptr = kbase + (size_t)(key0 + wave * 16) * DD;
            v8f s = {};
            v16bf bcur = load_B(kptr, DD);
            for (int kk = 0; kk < DD - 32; kk += 32) {
                v16bf a  = load_A(Qs + kk, DD);          // LDS
                v16bf bn = load_B(kptr + kk + 32, DD);   // global (L2-resident)
                s = WMMA_BF16(a, bcur, s);
                bcur = bn;
            }
            {
                v16bf a = load_A(Qs + DD - 32, DD);
                s = WMMA_BF16(a, bcur, s);
            }
#pragma unroll
            for (int r = 0; r < 8; ++r)
                Ss[(r + 8 * h) * 128 + wave * 16 + cl] = s[r];
        }
        __syncthreads();
        // --- cooperative online softmax (base-2; scale folded into Q)
        {
            const int row = t >> 4, c = t & 15;
            float mx = NEG_INF;
#pragma unroll
            for (int j = 0; j < 8; ++j) mx = fmaxf(mx, Ss[row * 128 + c + 16 * j]);
            pm[row][c] = mx;
        }
        __syncthreads();
        if (t < 16) {
            float mx = m_run[t];
#pragma unroll
            for (int c = 0; c < 16; ++c) mx = fmaxf(mx, pm[t][c]);
            m_new[t] = mx;
            corr[t]  = exp2f(m_run[t] - mx);
        }
        __syncthreads();
        {
            const int row = t >> 4, c = t & 15;
            const float m = m_new[row];
            float sum = 0.0f;
#pragma unroll
            for (int j = 0; j < 8; ++j) {
                float p = exp2f(Ss[row * 128 + c + 16 * j] - m);
                Ps[row * 128 + c + 16 * j] = f2bf(p);
                sum += p;
            }
            pl[row][c] = sum;
        }
        // rescale this wave's O accumulators (corr stable between barriers)
#pragma unroll
        for (int r = 0; r < 8; ++r) {
            const float c2 = corr[r + 8 * h];
#pragma unroll
            for (int j = 0; j < 8; ++j) o[j][r] *= c2;
        }
        __syncthreads();
        if (t < 16) {
            float s16 = 0.0f;
#pragma unroll
            for (int c = 0; c < 16; ++c) s16 += pl[t][c];
            l_run[t] = l_run[t] * corr[t] + s16;
            m_run[t] = m_new[t];
        }
        // --- O += P(16x128) @ V(128 x this wave's 128 D-cols), B pipelined
        {
            v16bf acur = {};
            v16bf pb = load_B(vtb + (size_t)(wave * 128) * NN + key0, NN);
#pragma unroll
            for (int u = 0; u < 32; ++u) {
                const int kk = u >> 3, j = u & 7;
                if (j == 0) acur = load_A(Ps + kk * 32, 128);   // LDS
                if (u + 1 < 32) {
                    const int un = u + 1;
                    v16bf bn = load_B(vtb + (size_t)(wave * 128 + 16 * (un & 7)) * NN
                                           + key0 + (un >> 3) * 32, NN);
                    o[j] = WMMA_BF16(acur, pb, o[j]);
                    pb = bn;
                } else {
                    o[j] = WMMA_BF16(acur, pb, o[j]);
                }
            }
        }
        __syncthreads();
    }
    // --- finalize: O /= l, store bf16
#pragma unroll
    for (int r = 0; r < 8; ++r) {
        const int row = r + 8 * h;
        const float inv = 1.0f / l_run[row];
        const int grow = m0 + row;
#pragma unroll
        for (int j = 0; j < 8; ++j) {
            const int col = wave * 128 + 16 * j + cl;
            oh[(size_t)grow * DD + col] = f2bf(o[j][r] * inv);
        }
    }
}

// ---------------- Kernel 4: output projection (fp32 out) ----------------
__global__ __launch_bounds__(128) void k_out(
    const bf16r* __restrict__ oh, const bf16r* __restrict__ woT,
    const float* __restrict__ bo, float* __restrict__ out)
{
    const int wave = threadIdx.x >> 5;
    const int lane = threadIdx.x & 31;
    const int m0 = (blockIdx.x * 4 + wave) * 32;
    const int n0 = blockIdx.y * 64;

    const bf16r* wb[4];
#pragma unroll
    for (int s = 0; s < 4; ++s) wb[s] = woT + (size_t)(n0 + 16 * s) * DD;

    v8f acc[4][2] = {};
    v16bf bcur = load_B(wb[0], DD);
    for (int kb = 0; kb < DD; kb += 32) {
        v16bf a0 = load_A(oh + (size_t)m0 * DD + kb, DD);
        v16bf a1 = load_A(oh + (size_t)(m0 + 16) * DD + kb, DD);
        if (kb + 32 < DD) {
            __builtin_prefetch(oh + (size_t)m0 * DD + kb + 32, 0, 3);
            __builtin_prefetch(oh + (size_t)(m0 + 16) * DD + kb + 32, 0, 3);
        }
        const int kbn = (kb + 32 < DD) ? kb + 32 : 0;
#pragma unroll
        for (int s = 0; s < 4; ++s) {
            const int sn = (s + 1) & 3;
            v16bf bn = load_B(wb[sn] + ((s == 3) ? kbn : kb), DD);
            acc[s][0] = WMMA_BF16(a0, bcur, acc[s][0]);
            acc[s][1] = WMMA_BF16(a1, bcur, acc[s][1]);
            bcur = bn;
        }
    }
    const int h = lane >> 4, cl = lane & 15;
#pragma unroll
    for (int j = 0; j < 4; ++j) {
        const int col = n0 + 16 * j + cl;
        const float bias = bo[col];
#pragma unroll
        for (int mt = 0; mt < 2; ++mt) {
#pragma unroll
            for (int r = 0; r < 8; ++r) {
                const int row = m0 + mt * 16 + r + 8 * h;
                out[(size_t)row * DD + col] = acc[j][mt][r] + bias;
            }
        }
    }
}

extern "C" void kernel_launch(void* const* d_in, const int* in_sizes, int n_in,
                              void* d_out, int out_size, void* d_ws, size_t ws_size,
                              hipStream_t stream)
{
    const float* x  = (const float*)d_in[0];
    const float* Wq = (const float*)d_in[1];
    const float* bq = (const float*)d_in[2];
    const float* Wk = (const float*)d_in[3];
    const float* bk = (const float*)d_in[4];
    const float* Wv = (const float*)d_in[5];
    const float* bv = (const float*)d_in[6];
    const float* Wo = (const float*)d_in[7];
    const float* bo = (const float*)d_in[8];

    const size_t MB32 = (size_t)SS * DD * sizeof(bf16r);   // 32 MB
    char* ws = (char*)d_ws;
    bf16r* xh  = (bf16r*)(ws);
    bf16r* qh  = (bf16r*)(ws + 1 * MB32);
    bf16r* kh  = (bf16r*)(ws + 2 * MB32);
    bf16r* vT  = (bf16r*)(ws + 3 * MB32);
    bf16r* oh  = (bf16r*)(ws + 4 * MB32);
    bf16r* wqT = (bf16r*)(ws + 5 * MB32);
    bf16r* wkT = wqT + (size_t)DD * DD;
    bf16r* wvT = wkT + (size_t)DD * DD;
    bf16r* woT = wvT + (size_t)DD * DD;

    // 1) precision conversion
    const int n4 = SS * DD / 4;
    k_cvt_x<<<(n4 + 255) / 256, 256, 0, stream>>>(x, xh, n4);
    const int wgrid = DD * DD / 256;
    k_cvt_wT<<<wgrid, 256, 0, stream>>>(Wq, wqT);
    k_cvt_wT<<<wgrid, 256, 0, stream>>>(Wk, wkT);
    k_cvt_wT<<<wgrid, 256, 0, stream>>>(Wv, wvT);
    k_cvt_wT<<<wgrid, 256, 0, stream>>>(Wo, woT);

    // 2) fused QKV projection: 512 m-tiles of 32 rows, 16 col-groups of 64
    dim3 g2(SS / 32 / 4, DD / 64);   // (128, 16)
    k_qkv<<<g2, 128, 0, stream>>>(xh, wqT, wkT, wvT, bq, bk, bv, qh, kh, vT);

    // 3) flash attention
    k_attn<<<BB * NN / 16, 256, 0, stream>>>(qh, kh, vT, oh);

    // 4) output projection
    k_out<<<g2, 128, 0, stream>>>(oh, woT, bo, (float*)d_out);
}
